// HimpNet_91079076479160
// MI455X (gfx1250) — matrix operand order repaired and checked by reading, not compiled, not run on hardware
//
#include <hip/hip_runtime.h>

// ---------------- problem constants (from reference) ----------------
#define NN 50000
#define EE 800000
#define CC 20000
#define FF 150000
#define GG 512
#define SS 64
#define HH 256
#define OO 128
#define LL 3

typedef __attribute__((ext_vector_type(16))) __bf16 v16bf;
typedef __attribute__((ext_vector_type(8)))  float  v8f;
typedef __attribute__((ext_vector_type(4)))  unsigned int v4u;

union Frag { v16bf v; v4u q[2]; };

__device__ __forceinline__ unsigned short f2bf(float f) {
  unsigned int u = __float_as_uint(f);
  unsigned int r = u + 0x7FFFu + ((u >> 16) & 1u);   // round-to-nearest-even
  return (unsigned short)(r >> 16);
}

// ---------------- elementwise helpers ----------------
__global__ void k_zero_f32(float* p, long n) {
  long i = (long)blockIdx.x * blockDim.x + threadIdx.x;
  for (; i < n; i += (long)gridDim.x * blockDim.x) p[i] = 0.0f;
}

__global__ void k_f32_to_bf16(const float* __restrict__ a, unsigned short* __restrict__ o, long n) {
  long i = (long)blockIdx.x * blockDim.x + threadIdx.x;
  if (i < n) o[i] = f2bf(a[i]);
}

// Pack a row-major f32 B[K,N] into bf16 WMMA fragment order:
// out[((tn*(K/32)+ks)*32 + lane)*16 + e] = B[ks*32 + 8*(lane>=16) + (e&7) + 16*(e>=8)][tn*16 + (lane&15)]
__global__ void k_packB(const float* __restrict__ B, unsigned short* __restrict__ out,
                        int K, int N) {
  long idx = (long)blockIdx.x * blockDim.x + threadIdx.x;
  if (idx >= (long)K * N) return;
  int e    = (int)(idx & 15);
  int lane = (int)((idx >> 4) & 31);
  long blk = idx >> 9;                      // = tn*(K/32) + ks
  int ksteps = K >> 5;
  int ks = (int)(blk % ksteps);
  int tn = (int)(blk / ksteps);
  int k = ks * 32 + ((lane >= 16) ? 8 : 0) + (e & 7) + ((e >= 8) ? 16 : 0);
  int n = tn * 16 + (lane & 15);
  out[idx] = f2bf(B[(long)k * N + n]);
}

// x[n,h] = sum_f atom_emb[f, x_atoms[n,f], h]
__global__ void k_atom_enc(const int* __restrict__ xa, const float* __restrict__ emb,
                           float* __restrict__ x) {
  long i = (long)blockIdx.x * blockDim.x + threadIdx.x;
  if (i >= (long)NN * HH) return;
  int n = (int)(i / HH), h = (int)(i % HH);
  float s = 0.0f;
#pragma unroll
  for (int f = 0; f < 9; ++f)
    s += emb[((long)f * 100 + xa[n * 9 + f]) * HH + h];
  x[i] = s;
}

// GINE message: msg = relu(x[src] + bond_emb sum); atomic scatter-add to agg[dst]
__global__ void k_edge_msg(const int* __restrict__ src, const int* __restrict__ dst,
                           const int* __restrict__ eattr, const float* __restrict__ bemb,
                           const float* __restrict__ x, float* __restrict__ agg) {
  int e = blockIdx.x;
  int h = threadIdx.x;                       // HH == blockDim.x
  if (e >= EE) return;
  int a0 = eattr[e * 3 + 0], a1 = eattr[e * 3 + 1], a2 = eattr[e * 3 + 2];
  float ea = bemb[(0 * 6 + a0) * HH + h] + bemb[(1 * 6 + a1) * HH + h] +
             bemb[(2 * 6 + a2) * HH + h];
  float m = x[(long)src[e] * HH + h] + ea;
  m = m > 0.0f ? m : 0.0f;
  atomicAdd(&agg[(long)dst[e] * HH + h], m);
}

// out_bf16 = bf16((1+eps[layer])*x + agg)
__global__ void k_gine_combine(const float* __restrict__ x, const float* __restrict__ agg,
                               const float* __restrict__ epsp, int layer,
                               unsigned short* __restrict__ o) {
  long i = (long)blockIdx.x * blockDim.x + threadIdx.x;
  if (i >= (long)NN * HH) return;
  float eps = 1.0f + epsp[layer];
  o[i] = f2bf(eps * x[i] + agg[i]);
}

// per-column sum / sumsq over M rows (N <= 512), LDS partials + global atomics
__global__ void k_colstats(const float* __restrict__ x, int M, int N,
                           float* __restrict__ sum, float* __restrict__ sumsq) {
  __shared__ float ls[512], lq[512];
  for (int c = threadIdx.x; c < N; c += blockDim.x) { ls[c] = 0.0f; lq[c] = 0.0f; }
  __syncthreads();
  long total = (long)M * N;
  long per = (total + gridDim.x - 1) / gridDim.x;
  long start = (long)blockIdx.x * per;
  long end = start + per; if (end > total) end = total;
  for (long i = start + threadIdx.x; i < end; i += blockDim.x) {
    float v = x[i];
    int c = (int)(i % N);
    atomicAdd(&ls[c], v);
    atomicAdd(&lq[c], v * v);
  }
  __syncthreads();
  for (int c = threadIdx.x; c < N; c += blockDim.x) {
    atomicAdd(&sum[c], ls[c]);
    atomicAdd(&sumsq[c], lq[c]);
  }
}

// y = relu(bn(in)); optional f32 and/or bf16 outputs
__global__ void k_bn_relu(const float* __restrict__ in, float* of32, unsigned short* ob16,
                          const float* __restrict__ g, const float* __restrict__ b,
                          const float* __restrict__ sum, const float* __restrict__ sumsq,
                          int M, int N) {
  long i = (long)blockIdx.x * blockDim.x + threadIdx.x;
  if (i >= (long)M * N) return;
  int c = (int)(i % N);
  float m = sum[c] / (float)M;
  float var = sumsq[c] / (float)M - m * m;
  float y = (in[i] - m) * rsqrtf(var + 1e-5f) * g[c] + b[c];
  y = y > 0.0f ? y : 0.0f;
  if (of32) of32[i] = y;
  if (ob16) ob16[i] = f2bf(y);
}

// acc[dst[f]] += x[src ? src[f] : f]; cnt[dst[f]] += 1
__global__ void k_seg_scatter(const int* __restrict__ src_idx, const int* __restrict__ dst_idx,
                              const float* __restrict__ x, float* __restrict__ acc,
                              float* __restrict__ cnt, int Fn) {
  int f = blockIdx.x;
  int h = threadIdx.x;                       // HH
  if (f >= Fn) return;
  int r = src_idx ? src_idx[f] : f;
  int c = dst_idx[f];
  atomicAdd(&acc[(long)c * HH + h], x[(long)r * HH + h]);
  if (h == 0) atomicAdd(&cnt[c], 1.0f);
}

// o = bf16(acc / max(cnt,1))
__global__ void k_segdiv_bf16(const float* __restrict__ acc, const float* __restrict__ cnt,
                              unsigned short* __restrict__ o, int M) {
  long i = (long)blockIdx.x * blockDim.x + threadIdx.x;
  if (i >= (long)M * HH) return;
  float c = cnt[i / HH];
  c = c < 1.0f ? 1.0f : c;
  o[i] = f2bf(acc[i] / c);
}

__global__ void k_add_relu(float* __restrict__ dst, const float* __restrict__ src, long n) {
  long i = (long)blockIdx.x * blockDim.x + threadIdx.x;
  if (i >= n) return;
  float v = src[i];
  dst[i] += (v > 0.0f ? v : 0.0f);
}

__global__ void k_relu_sum_bf16(const float* __restrict__ a, const float* __restrict__ b,
                                unsigned short* __restrict__ o, long n) {
  long i = (long)blockIdx.x * blockDim.x + threadIdx.x;
  if (i >= n) return;
  float v = a[i] + b[i];
  o[i] = f2bf(v > 0.0f ? v : 0.0f);
}

// ---------------- WMMA bf16 GEMM ----------------
// C[M,N] = A[M,K] @ B[K,N] + bias. A row-major bf16; B pre-packed fragment order.
// One wave computes a 16x32 output strip (2 N-tiles) to reuse each A fragment.
template <int K>
__global__ void k_gemm_bf16(const unsigned short* __restrict__ A,
                            const unsigned short* __restrict__ Bp,
                            const float* __restrict__ bias, float* __restrict__ C,
                            int M, int N) {
  constexpr int KS = K >> 5;                 // K-steps of 32
  int wave = (int)(((long)blockIdx.x * blockDim.x + threadIdx.x) >> 5);
  int lane = threadIdx.x & 31;
  int stripsN = N >> 5;
  int strips = (M >> 4) * stripsN;
  if (wave >= strips) return;                // wave-uniform exit (EXEC stays all-1)
  int tm = wave / stripsN, tn2 = wave % stripsN;

  int hi = (lane >= 16) ? 1 : 0;
  int m_load = tm * 16 + (lane & 15);        // A: lane = row M
  const unsigned short* arow = A + (long)m_load * K + (hi ? 8 : 0);
  // packed B: per tile, KS fragments of 512 elems; per lane, 16 contiguous bf16
  const unsigned short* bp0 = Bp + ((long)(2 * tn2) * KS * 512) + lane * 16;
  const unsigned short* bp1 = bp0 + (long)KS * 512;

  v8f acc0 = {}, acc1 = {};
#pragma unroll 4
  for (int ks = 0; ks < KS; ++ks) {
    __builtin_prefetch(arow + ks * 32 + 32, 0, 3);
    Frag a, b0, b1;
    a.q[0]  = *(const v4u*)(arow + ks * 32);        // K=base..base+7
    a.q[1]  = *(const v4u*)(arow + ks * 32 + 16);   // K=base+16..base+23
    b0.q[0] = *(const v4u*)(bp0 + ks * 512);
    b0.q[1] = *(const v4u*)(bp0 + ks * 512 + 8);
    b1.q[0] = *(const v4u*)(bp1 + ks * 512);
    b1.q[1] = *(const v4u*)(bp1 + ks * 512 + 8);
    acc0 = __builtin_amdgcn_wmma_f32_16x16x32_bf16(false, a.v, false, b0.v,
                                                   (short)0, acc0, false, false);
    acc1 = __builtin_amdgcn_wmma_f32_16x16x32_bf16(false, a.v, false, b1.v,
                                                   (short)0, acc1, false, false);
  }

  int col0 = tn2 * 32 + (lane & 15);
  int col1 = col0 + 16;
  float bv0 = bias ? bias[col0] : 0.0f;
  float bv1 = bias ? bias[col1] : 0.0f;
#pragma unroll
  for (int r = 0; r < 8; ++r) {              // C/D: VGPR r -> row r (+8 for hi lanes)
    long row = tm * 16 + r + (hi ? 8 : 0);
    C[row * N + col0] = acc0[r] + bv0;
    C[row * N + col1] = acc1[r] + bv1;
  }
}

// ---------------- host orchestration ----------------
extern "C" void kernel_launch(void* const* d_in, const int* in_sizes, int n_in,
                              void* d_out, int out_size, void* d_ws, size_t ws_size,
                              hipStream_t stream) {
  (void)in_sizes; (void)n_in; (void)out_size; (void)ws_size;
  const int*   x_atoms   = (const int*)d_in[0];
  const int*   edge_attr = (const int*)d_in[1];
  const int*   edge_index= (const int*)d_in[2];
  const int*   frag_row  = (const int*)d_in[3];
  const int*   frag_col  = (const int*)d_in[4];
  const int*   batch     = (const int*)d_in[5];
  const int*   frag_batch= (const int*)d_in[6];
  const float* fragments = (const float*)d_in[7];
  const float* atom_emb  = (const float*)d_in[8];
  const float* cenc_w    = (const float*)d_in[9];
  const float* cenc_b    = (const float*)d_in[10];
  const float* bond_emb  = (const float*)d_in[11];
  const float* eps_atom  = (const float*)d_in[12];
  const float* conv_w1   = (const float*)d_in[13];
  const float* conv_b1   = (const float*)d_in[14];
  const float* conv_bn_g = (const float*)d_in[15];
  const float* conv_bn_b = (const float*)d_in[16];
  const float* conv_w2   = (const float*)d_in[17];
  const float* conv_b2   = (const float*)d_in[18];
  const float* atom_bn_g = (const float*)d_in[19];
  const float* atom_bn_b = (const float*)d_in[20];
  const float* a2c_w     = (const float*)d_in[21];
  const float* a2c_b     = (const float*)d_in[22];
  const float* c2a_w     = (const float*)d_in[23];
  const float* c2a_b     = (const float*)d_in[24];
  const float* cl_bn_g   = (const float*)d_in[25];
  const float* cl_bn_b   = (const float*)d_in[26];
  const float* alin_w    = (const float*)d_in[27];
  const float* alin_b    = (const float*)d_in[28];
  const float* clin_w    = (const float*)d_in[29];
  const float* clin_b    = (const float*)d_in[30];
  const float* lin_w     = (const float*)d_in[31];
  const float* lin_b     = (const float*)d_in[32];
  float* out = (float*)d_out;

  const int* e_src = edge_index;            // edge_index[0, :]
  const int* e_dst = edge_index + EE;       // edge_index[1, :]

  // ---- carve workspace ----
  char* wp = (char*)d_ws;
  auto carve = [&](size_t bytes) { void* p = wp; wp += (bytes + 255) & ~(size_t)255; return p; };
  float*          x    = (float*)carve((size_t)NN * HH * 4);
  float*          xc   = (float*)carve((size_t)CC * HH * 4);
  float*          F1   = (float*)carve((size_t)NN * 2 * HH * 4);   // h (N x 512)
  float*          F2   = (float*)carve((size_t)NN * HH * 4);       // agg / gemm2 out / tmpN
  unsigned short* B1   = (unsigned short*)carve((size_t)NN * HH * 2);
  unsigned short* B2   = (unsigned short*)carve((size_t)NN * 2 * HH * 2);
  float*          C1   = (float*)carve((size_t)CC * HH * 4);
  unsigned short* Cb   = (unsigned short*)carve((size_t)CC * HH * 2);
  float*          cnt  = (float*)carve((size_t)NN * 4);            // reused for C/G counts
  float*          ssum = (float*)carve(512 * 4);
  float*          ssq  = (float*)carve(512 * 4);
  float*          poolA= (float*)carve((size_t)GG * HH * 4);
  float*          xg   = (float*)carve((size_t)GG * HH * 4);
  float*          xcg  = (float*)carve((size_t)GG * HH * 4);
  unsigned short* pb16 = (unsigned short*)carve((size_t)GG * HH * 2);
  // bf16 copies: fragments (A-matrix) + packed B weights
  unsigned short* fragb= (unsigned short*)carve((size_t)CC * SS * 2);
  unsigned short* cencb= (unsigned short*)carve((size_t)SS * HH * 2);
  unsigned short* w1b  = (unsigned short*)carve((size_t)LL * HH * 2 * HH * 2);
  unsigned short* w2b  = (unsigned short*)carve((size_t)LL * 2 * HH * HH * 2);
  unsigned short* a2cb = (unsigned short*)carve((size_t)LL * HH * HH * 2);
  unsigned short* c2ab = (unsigned short*)carve((size_t)LL * HH * HH * 2);
  unsigned short* alinb= (unsigned short*)carve((size_t)HH * HH * 2);
  unsigned short* clinb= (unsigned short*)carve((size_t)HH * HH * 2);
  unsigned short* linb = (unsigned short*)carve((size_t)HH * OO * 2);

  auto gridN = [](long n) { return (unsigned)((n + 255) / 256); };
  auto zero  = [&](float* p, long n) { k_zero_f32<<<gridN(n) < 4096u ? gridN(n) : 4096u, 256, 0, stream>>>(p, n); };
  auto tobf  = [&](const float* a, unsigned short* o, long n) { k_f32_to_bf16<<<gridN(n), 256, 0, stream>>>(a, o, n); };
  auto packB = [&](const float* B, unsigned short* o, int K, int N) {
    k_packB<<<gridN((long)K * N), 256, 0, stream>>>(B, o, K, N);
  };
  auto gemm  = [&](const unsigned short* A, const unsigned short* Bp, const float* bias,
                   float* Cm, int M, int N, int K) {
    long waves = (long)(M / 16) * (N / 32);
    unsigned grid = gridN(waves * 32);
    switch (K) {
      case 64:  k_gemm_bf16<64> <<<grid, 256, 0, stream>>>(A, Bp, bias, Cm, M, N); break;
      case 256: k_gemm_bf16<256><<<grid, 256, 0, stream>>>(A, Bp, bias, Cm, M, N); break;
      case 512: k_gemm_bf16<512><<<grid, 256, 0, stream>>>(A, Bp, bias, Cm, M, N); break;
    }
  };
  auto stats = [&](const float* p, int M, int N) {
    zero(ssum, N); zero(ssq, N);
    k_colstats<<<256, 256, 0, stream>>>(p, M, N, ssum, ssq);
  };

  // ---- weight conversions / packing (deterministic, cheap) ----
  tobf(fragments, fragb, (long)CC * SS);                       // A matrix: row-major bf16
  packB(cenc_w, cencb, SS, HH);
  for (int i = 0; i < LL; ++i) {
    packB(conv_w1 + (long)i * HH * 2 * HH, w1b + (long)i * HH * 2 * HH, HH, 2 * HH);
    packB(conv_w2 + (long)i * 2 * HH * HH, w2b + (long)i * 2 * HH * HH, 2 * HH, HH);
    packB(a2c_w   + (long)i * HH * HH,     a2cb + (long)i * HH * HH,    HH, HH);
    packB(c2a_w   + (long)i * HH * HH,     c2ab + (long)i * HH * HH,    HH, HH);
  }
  packB(alin_w, alinb, HH, HH);
  packB(clin_w, clinb, HH, HH);
  packB(lin_w, linb, HH, OO);

  // ---- encoders ----
  k_atom_enc<<<gridN((long)NN * HH), 256, 0, stream>>>(x_atoms, atom_emb, x);
  gemm(fragb, cencb, cenc_b, xc, CC, HH, SS);          // x_clique = fragments @ W + b

  // ---- layers ----
  for (int i = 0; i < LL; ++i) {
    // GINE message + aggregate
    zero(F2, (long)NN * HH);
    k_edge_msg<<<EE, HH, 0, stream>>>(e_src, e_dst, edge_attr,
                                      bond_emb + (long)i * 3 * 6 * HH, x, F2);
    k_gine_combine<<<gridN((long)NN * HH), 256, 0, stream>>>(x, F2, eps_atom, i, B1);
    // MLP: h = relu(bn(h @ W1 + b1)); x = relu(bn(h @ W2 + b2))
    gemm(B1, w1b + (long)i * HH * 2 * HH, conv_b1 + (long)i * 2 * HH, F1, NN, 2 * HH, HH);
    stats(F1, NN, 2 * HH);
    k_bn_relu<<<gridN((long)NN * 2 * HH), 256, 0, stream>>>(
        F1, nullptr, B2, conv_bn_g + (long)i * 2 * HH, conv_bn_b + (long)i * 2 * HH,
        ssum, ssq, NN, 2 * HH);
    gemm(B2, w2b + (long)i * 2 * HH * HH, conv_b2 + (long)i * HH, F2, NN, HH, 2 * HH);
    stats(F2, NN, HH);
    k_bn_relu<<<gridN((long)NN * HH), 256, 0, stream>>>(
        F2, x, nullptr, atom_bn_g + (long)i * HH, atom_bn_b + (long)i * HH,
        ssum, ssq, NN, HH);
    // atoms -> cliques
    zero(C1, (long)CC * HH); zero(cnt, CC);
    k_seg_scatter<<<FF, HH, 0, stream>>>(frag_row, frag_col, x, C1, cnt, FF);
    k_segdiv_bf16<<<gridN((long)CC * HH), 256, 0, stream>>>(C1, cnt, Cb, CC);
    gemm(Cb, a2cb + (long)i * HH * HH, a2c_b + (long)i * HH, C1, CC, HH, HH);
    k_add_relu<<<gridN((long)CC * HH), 256, 0, stream>>>(xc, C1, (long)CC * HH);
    stats(xc, CC, HH);
    k_bn_relu<<<gridN((long)CC * HH), 256, 0, stream>>>(
        xc, xc, nullptr, cl_bn_g + (long)i * HH, cl_bn_b + (long)i * HH,
        ssum, ssq, CC, HH);
    // cliques -> atoms
    zero(F2, (long)NN * HH); zero(cnt, NN);
    k_seg_scatter<<<FF, HH, 0, stream>>>(frag_col, frag_row, xc, F2, cnt, FF);
    k_segdiv_bf16<<<gridN((long)NN * HH), 256, 0, stream>>>(F2, cnt, B1, NN);
    gemm(B1, c2ab + (long)i * HH * HH, c2a_b + (long)i * HH, F2, NN, HH, HH);
    k_add_relu<<<gridN((long)NN * HH), 256, 0, stream>>>(x, F2, (long)NN * HH);
  }

  // ---- readout ----
  zero(poolA, (long)GG * HH); zero(cnt, GG);
  k_seg_scatter<<<NN, HH, 0, stream>>>(nullptr, batch, x, poolA, cnt, NN);
  k_segdiv_bf16<<<gridN((long)GG * HH), 256, 0, stream>>>(poolA, cnt, pb16, GG);
  gemm(pb16, alinb, alin_b, xg, GG, HH, HH);

  zero(poolA, (long)GG * HH); zero(cnt, GG);
  k_seg_scatter<<<CC, HH, 0, stream>>>(nullptr, frag_batch, xc, poolA, cnt, CC);
  k_segdiv_bf16<<<gridN((long)GG * HH), 256, 0, stream>>>(poolA, cnt, pb16, GG);
  gemm(pb16, clinb, clin_b, xcg, GG, HH, HH);

  k_relu_sum_bf16<<<gridN((long)GG * HH), 256, 0, stream>>>(xg, xcg, pb16, (long)GG * HH);
  gemm(pb16, linb, lin_b, out, GG, OO, HH);
}
